// HyperGraphBasicConvolution_41205916237980
// MI455X (gfx1250) — compile-verified, matrix operand order checked by reading, not run on hardware
//
#include <hip/hip_runtime.h>
#include <hip/hip_bf16.h>
#include <stdint.h>

// ---------------------------------------------------------------------------
// Problem constants (from reference): U=I=16384, G=4096, N=36864, M=16, D=128
// ---------------------------------------------------------------------------
#define DIMU 16384
#define DIMI 16384
#define DIMG 4096
#define DIMN 36864
#define DIMD 128
#define BIAS_C (-0.73f)

typedef __attribute__((ext_vector_type(16))) __bf16 v16bf;
typedef __attribute__((ext_vector_type(8)))  float  v8f;

union FragBF { v16bf v; uint4 q[2]; };

__device__ __forceinline__ unsigned short f2bf(float f) {
  unsigned int u = __float_as_uint(f);
  unsigned int r = u + 0x7FFFu + ((u >> 16) & 1u);   // round-to-nearest-even
  return (unsigned short)(r >> 16);
}
__device__ __forceinline__ float bf2f(unsigned short h) {
  return __uint_as_float(((unsigned int)h) << 16);
}

// ---------------------------------------------------------------------------
// Zero-fill kernel (outputs/accumulators are atomically accumulated)
// ---------------------------------------------------------------------------
__global__ void zero_f32_kernel(float* __restrict__ p, long n4) {
  long i = (long)blockIdx.x * blockDim.x + threadIdx.x;
  float4 z = make_float4(0.f, 0.f, 0.f, 0.f);
  for (; i < n4; i += (long)gridDim.x * blockDim.x)
    ((float4*)p)[i] = z;
}

// ---------------------------------------------------------------------------
// Generic C[M,128] += A[M,K] @ B[K,128], fp32 in / fp32 out, split-bf16 WMMA.
//   A = Ahi + Alo (bf16 hi + bf16 residual); same for B.
//   acc += Ah*Bh + Ah*Bl + Al*Bh   (~fp32 accuracy, 3x bf16 compute which is
//   free relative to the 23.3 TB/s memory floor at 64 FLOP/byte intensity)
// Tiling: BM=128 rows x BN=128 (all of D) x BK=32, 256 thr = 8 waves,
// wave w owns rows [w*16, w*16+16) -> 8 WMMA C-tiles (64 acc VGPRs/lane).
// K split over blockIdx.y (length KC each), accumulated with atomics.
// ---------------------------------------------------------------------------
#define BM 128
#define BK 32
#define LS 40   // ushorts per LDS row (80 B): padded -> conflict-free b128 reads

__global__ __launch_bounds__(256) void gemm128_splitbf16_kernel(
    const float* __restrict__ A, const float* __restrict__ B,
    float* __restrict__ C, int K, int KC)
{
  __shared__ unsigned short AsH[BM * LS];
  __shared__ unsigned short AsL[BM * LS];
  __shared__ unsigned short BsH[DIMD * LS];
  __shared__ unsigned short BsL[DIMD * LS];

  const int tid    = threadIdx.x;
  const int lane   = tid & 31;
  const int wave   = tid >> 5;                 // 0..7
  const int lrow   = lane & 15;                // 0..15
  const int khalf  = (lane >> 4) * 8;          // 0 or 8 (A/B K packing, ISA 7.12.2)
  const int rowT   = blockIdx.x * BM;
  const int k0     = blockIdx.y * KC;
  const int kEnd   = k0 + KC;

  v8f acc[8];
#pragma unroll
  for (int n = 0; n < 8; ++n) acc[n] = (v8f){0.f,0.f,0.f,0.f,0.f,0.f,0.f,0.f};

  for (int kk = k0; kk < kEnd; kk += BK) {
    // ---- stage A tile: 128 rows x 32 fp32 -> bf16 hi/lo in LDS (row-major)
#pragma unroll
    for (int j = 0; j < 4; ++j) {
      int idx = tid + j * 256;                 // 0..1023
      int r   = idx >> 3;                      // row 0..127
      int c4  = (idx & 7) * 4;                 // k offset 0..28
      float4 f = *(const float4*)(A + (size_t)(rowT + r) * K + kk + c4);
      unsigned short h0 = f2bf(f.x), h1 = f2bf(f.y), h2 = f2bf(f.z), h3 = f2bf(f.w);
      unsigned short l0 = f2bf(f.x - bf2f(h0)), l1 = f2bf(f.y - bf2f(h1));
      unsigned short l2 = f2bf(f.z - bf2f(h2)), l3 = f2bf(f.w - bf2f(h3));
      *(uint2*)&AsH[r * LS + c4] = make_uint2((unsigned)h0 | ((unsigned)h1 << 16),
                                              (unsigned)h2 | ((unsigned)h3 << 16));
      *(uint2*)&AsL[r * LS + c4] = make_uint2((unsigned)l0 | ((unsigned)l1 << 16),
                                              (unsigned)l2 | ((unsigned)l3 << 16));
    }
    // ---- stage B tile: 32 k-rows x 128 cols fp32 -> LDS transposed [n][k]
#pragma unroll
    for (int j = 0; j < 4; ++j) {
      int idx = tid + j * 256;
      int kr  = idx >> 5;                      // k-row 0..31
      int c4  = (idx & 31) * 4;                // col 0..124
      float4 f = *(const float4*)(B + (size_t)(kk + kr) * DIMD + c4);
      float fv[4] = {f.x, f.y, f.z, f.w};
#pragma unroll
      for (int t = 0; t < 4; ++t) {
        unsigned short h = f2bf(fv[t]);
        BsH[(c4 + t) * LS + kr] = h;
        BsL[(c4 + t) * LS + kr] = f2bf(fv[t] - bf2f(h));
      }
    }
    __syncthreads();

    // ---- A fragments for this wave's 16 rows
    FragBF aH, aL;
    const int arow = wave * 16 + lrow;
    aH.q[0] = *(const uint4*)&AsH[arow * LS + khalf];
    aH.q[1] = *(const uint4*)&AsH[arow * LS + khalf + 16];
    aL.q[0] = *(const uint4*)&AsL[arow * LS + khalf];
    aL.q[1] = *(const uint4*)&AsL[arow * LS + khalf + 16];

#pragma unroll
    for (int n = 0; n < 8; ++n) {
      FragBF bH, bL;
      const int bn = n * 16 + lrow;
      bH.q[0] = *(const uint4*)&BsH[bn * LS + khalf];
      bH.q[1] = *(const uint4*)&BsH[bn * LS + khalf + 16];
      bL.q[0] = *(const uint4*)&BsL[bn * LS + khalf];
      bL.q[1] = *(const uint4*)&BsL[bn * LS + khalf + 16];
      acc[n] = __builtin_amdgcn_wmma_f32_16x16x32_bf16(
          false, aH.v, false, bH.v, (short)0, acc[n], false, false);
      acc[n] = __builtin_amdgcn_wmma_f32_16x16x32_bf16(
          false, aH.v, false, bL.v, (short)0, acc[n], false, false);
      acc[n] = __builtin_amdgcn_wmma_f32_16x16x32_bf16(
          false, aL.v, false, bH.v, (short)0, acc[n], false, false);
    }
    __syncthreads();
  }

  // ---- C layout (ISA 7.12.2): VGPR v: lanes 0-15 -> M=v, lanes 16-31 -> M=v+8
  const int m0 = rowT + wave * 16 + ((lane >> 4) << 3);
#pragma unroll
  for (int n = 0; n < 8; ++n) {
#pragma unroll
    for (int v = 0; v < 8; ++v) {
      atomicAdd(&C[(size_t)(m0 + v) * DIMD + n * 16 + lrow], acc[n][v]);
    }
  }
}

// ---------------------------------------------------------------------------
// Per-group glue: classifier (argmax over size-1 axis -> member 0), selection,
// elementwise product, and the tiny [G,384] @ [384,128] W_agg projection.
// One block (128 threads) per group; exact fp32.
// ---------------------------------------------------------------------------
__global__ __launch_bounds__(128) void group_msg_kernel(
    const float* __restrict__ user_msg, const float* __restrict__ item_msg,
    const float* __restrict__ group_emb, const float* __restrict__ weight,
    const float* __restrict__ gmemb,     const float* __restrict__ W_agg,
    const float* __restrict__ b_agg,     const float* __restrict__ W_cls,
    const float* __restrict__ b_cls,
    float* __restrict__ msg_out, float* __restrict__ pc_out)
{
  const int g = blockIdx.x;
  const int d = threadIdx.x;
  __shared__ float cbuf[3 * DIMD];

  const float w  = weight[g];
  const float s0 = w * W_cls[0] + b_cls[0];
  const float s1 = w * W_cls[1] + b_cls[1] + BIAS_C;
  const int   pc = (s1 > s0) ? 1 : 0;          // argmax picks first index on tie

  const float im = item_msg[(size_t)g * DIMD + d];
  // max_idx == 0 always (weight axis-1 has extent 1)
  const float rv = pc ? gmemb[(size_t)g * 16 * DIMD + d]
                      : user_msg[(size_t)g * DIMD + d];
  cbuf[d]            = rv;
  cbuf[DIMD + d]     = im;
  cbuf[2 * DIMD + d] = im * group_emb[(size_t)g * DIMD + d];
  __syncthreads();

  float accv = b_agg[d];
  const float* wrow = W_agg + (size_t)d * (3 * DIMD);   // W_agg[d, :]
#pragma unroll 4
  for (int j = 0; j < 3 * DIMD; ++j) accv = fmaf(cbuf[j], wrow[j], accv);

  msg_out[(size_t)g * DIMD + d] = accv;
  if (d == 0) pc_out[g] = (float)pc;
}

// ---------------------------------------------------------------------------
// Launch
// ---------------------------------------------------------------------------
extern "C" void kernel_launch(void* const* d_in, const int* in_sizes, int n_in,
                              void* d_out, int out_size, void* d_ws, size_t ws_size,
                              hipStream_t stream) {
  const float* user_emb   = (const float*)d_in[0];
  const float* item_emb   = (const float*)d_in[1];
  const float* group_emb  = (const float*)d_in[2];
  const float* user_hg    = (const float*)d_in[3];
  const float* item_hg    = (const float*)d_in[4];
  const float* full_hyper = (const float*)d_in[5];
  const float* weight     = (const float*)d_in[6];
  const float* gmemb      = (const float*)d_in[7];
  const float* W_agg      = (const float*)d_in[8];
  const float* b_agg      = (const float*)d_in[9];
  const float* W_cls      = (const float*)d_in[10];
  const float* b_cls      = (const float*)d_in[11];

  float* norm_emb = (float*)d_out;                               // [N, D]
  float* msg      = norm_emb + (size_t)DIMN * DIMD;              // [G, D]
  float* pc       = msg + (size_t)DIMG * DIMD;                   // [G]

  // Stage user_msg/item_msg in the norm_emb region (re-zeroed before final GEMM)
  float* Pu = norm_emb;
  float* Pi = norm_emb + (size_t)DIMG * DIMD;

  // 1) zero the two hyper-GEMM accumulators
  zero_f32_kernel<<<1024, 256, 0, stream>>>(Pu, (long)(2 * DIMG * DIMD) / 4);

  // 2) user_msg = user_hg @ user_emb   (K=16384 split 8 ways)
  gemm128_splitbf16_kernel<<<dim3(DIMG / BM, 8), 256, 0, stream>>>(
      user_hg, user_emb, Pu, DIMU, DIMU / 8);
  // 3) item_msg = item_hg @ item_emb
  gemm128_splitbf16_kernel<<<dim3(DIMG / BM, 8), 256, 0, stream>>>(
      item_hg, item_emb, Pi, DIMI, DIMI / 8);

  // 4) per-group selection + W_agg projection -> msg, predicted classes
  group_msg_kernel<<<DIMG, DIMD, 0, stream>>>(
      Pu, Pi, group_emb, weight, gmemb, W_agg, b_agg, W_cls, b_cls, msg, pc);

  // 5) zero norm_emb (also wipes the staged Pu/Pi, no longer needed)
  zero_f32_kernel<<<4608, 256, 0, stream>>>(norm_emb, (long)DIMN * DIMD / 4);

  // 6) norm_emb = full_hyper @ msg   (K=4096 split 4 ways)
  gemm128_splitbf16_kernel<<<dim3(DIMN / BM, 4), 256, 0, stream>>>(
      full_hyper, msg, norm_emb, DIMG, DIMG / 4);
}
